// Model_67027259622099
// MI455X (gfx1250) — matrix operand order repaired and checked by reading, not compiled, not run on hardware
//
#include <hip/hip_runtime.h>

#define EPS 1e-4f

typedef __attribute__((ext_vector_type(16))) __bf16 v16bf;
typedef __attribute__((ext_vector_type(8)))  float  v8f;

struct alignas(16) U4 { unsigned int a, b, c, d; };
union Frag { v16bf v; U4 q[2]; };

__device__ __forceinline__ unsigned short f2bf(float f) {
    unsigned int u = __float_as_uint(f);
    u += 0x7fffu + ((u >> 16) & 1u);   // round-to-nearest-even
    return (unsigned short)(u >> 16);
}

// ---------------- elementwise helpers ----------------

__global__ void k_zero(float* p, int n) {
    int i = blockIdx.x * blockDim.x + threadIdx.x;
    if (i < n) p[i] = 0.f;
}

__global__ void k_maskf(const int* __restrict__ m, float* __restrict__ o, int n) {
    int i = blockIdx.x * blockDim.x + threadIdx.x;
    if (i < n) o[i] = (float)m[i];
}

// fp32 [taps][Cin][Cout] -> bf16 [taps][CoutPad][Cin] (zero pad co>=Cout)
__global__ void k_wconv(const float* __restrict__ w, unsigned short* __restrict__ o,
                        int taps, int Cin, int Cout, int CoutPad) {
    size_t i = (size_t)blockIdx.x * blockDim.x + threadIdx.x;
    size_t total = (size_t)taps * CoutPad * Cin;
    if (i >= total) return;
    int ci = (int)(i % Cin);
    size_t t = i / Cin;
    int co = (int)(t % CoutPad);
    int tap = (int)(t / CoutPad);
    float v = (co < Cout) ? w[((size_t)tap * Cin + ci) * Cout + co] : 0.f;
    o[i] = f2bf(v);
}

// input submanifold conv: Cin=1, h = conv(x*mask) * mask, fp32
__global__ void k_input_conv(const float* __restrict__ x, const float* __restrict__ mf,
                             const float* __restrict__ w, float* __restrict__ out,
                             int B, int S, int Cout) {
    size_t i = (size_t)blockIdx.x * blockDim.x + threadIdx.x;
    size_t total = (size_t)B * S * S * S * Cout;
    if (i >= total) return;
    int co = (int)(i % Cout);
    size_t s = i / Cout;
    int xo = (int)(s % S); size_t t = s / S;
    int yo = (int)(t % S); t /= S;
    int zo = (int)(t % S); int b = (int)(t / S);
    float acc = 0.f;
    for (int tap = 0; tap < 27; ++tap) {
        int dz = tap / 9, dy = (tap / 3) % 3, dx = tap % 3;
        int zi = zo + dz - 1, yi = yo + dy - 1, xi = xo + dx - 1;
        if ((unsigned)zi < (unsigned)S && (unsigned)yi < (unsigned)S && (unsigned)xi < (unsigned)S) {
            size_t ii = (((size_t)b * S + zi) * S + yi) * S + xi;
            acc += x[ii] * mf[ii] * w[tap * Cout + co];
        }
    }
    out[i] = acc * mf[s];
}

// masked BN statistics: stats[0]=n, stats[1..C]=sum, stats[1+C..2C]=sumsq
// blockDim=256; thread owns channels (threadIdx.x, threadIdx.x+256); private
// accumulation over grid-strided sites -> one global atomic per block/channel.
__global__ void k_bn_stats(const float* __restrict__ x, const float* __restrict__ mask,
                           float* __restrict__ stats, int M, int C, int rowStride) {
    const int c0 = threadIdx.x;
    const int c1 = threadIdx.x + 256;
    const bool hasA = c0 < C;
    const bool hasB = c1 < C;
    float s0a = 0.f, s1a = 0.f, s0b = 0.f, s1b = 0.f, cnt = 0.f;
    for (int site = blockIdx.x; site < M; site += gridDim.x) {
        float m = mask[site];
        if (threadIdx.x == 0) cnt += m;
        const float* row = x + (size_t)site * rowStride;
        if (hasA) { float v = row[c0] * m; s0a += v; s1a += v * v; }
        if (hasB) { float v = row[c1] * m; s0b += v; s1b += v * v; }
    }
    if (hasA) { atomicAdd(&stats[1 + c0], s0a); atomicAdd(&stats[1 + C + c0], s1a); }
    if (hasB) { atomicAdd(&stats[1 + c1], s0b); atomicAdd(&stats[1 + C + c1], s1b); }
    if (threadIdx.x == 0) atomicAdd(&stats[0], cnt);
}

// masked BN + ReLU, emit bf16
__global__ void k_bn_relu(const float* __restrict__ x, const float* __restrict__ mask,
                          const float* __restrict__ stats, const float* __restrict__ g,
                          const float* __restrict__ bt, unsigned short* __restrict__ out,
                          int M, int C, int rowStride) {
    size_t i = (size_t)blockIdx.x * blockDim.x + threadIdx.x;
    size_t total = (size_t)M * C;
    if (i >= total) return;
    int c = (int)(i % C);
    size_t site = i / C;
    float n = stats[0];
    float mean = stats[1 + c] / n;
    float var = stats[1 + C + c] / n - mean * mean;
    float y = (x[site * rowStride + c] - mean) * rsqrtf(var + EPS) * g[c] + bt[c];
    y = fmaxf(y, 0.f) * mask[site];
    out[i] = f2bf(y);
}

// coarse mask: active iff any fine-mask voxel in 3^3 receptive field (stride 2, VALID)
__global__ void k_cmask(const float* __restrict__ fm, float* __restrict__ cm,
                        int B, int Sf, int Sc) {
    int i = blockIdx.x * blockDim.x + threadIdx.x;
    int total = B * Sc * Sc * Sc;
    if (i >= total) return;
    int xo = i % Sc; int t = i / Sc;
    int yo = t % Sc; t /= Sc;
    int zo = t % Sc; int b = t / Sc;
    float any = 0.f;
    for (int dz = 0; dz < 3; ++dz)
        for (int dy = 0; dy < 3; ++dy)
            for (int dx = 0; dx < 3; ++dx) {
                int zi = 2 * zo + dz, yi = 2 * yo + dy, xi = 2 * xo + dx;
                any += fm[(((size_t)b * Sf + zi) * Sf + yi) * Sf + xi];
            }
    cm[i] = (any > 0.f) ? 1.f : 0.f;
}

__device__ __forceinline__ int upcand(int f, int Sc, int* js) {
    int n = 0;
    if (f & 1) { int j = (f - 1) >> 1; if (j < Sc) js[n++] = j; }
    else { int j = f >> 1; if (j < Sc) js[n++] = j; if (j - 1 >= 0) js[n++] = j - 1; }
    return n;
}

// unpool (k=3, s=2): fine site accumulates covering coarse features, then mask
__global__ void k_unpool(const float* __restrict__ coarse, int csStride, int cC,
                         float* __restrict__ fine, int fsStride, int fOff,
                         const float* __restrict__ fm, int B, int Sc, int Sf) {
    size_t i = (size_t)blockIdx.x * blockDim.x + threadIdx.x;
    size_t total = (size_t)B * Sf * Sf * Sf * cC;
    if (i >= total) return;
    int c = (int)(i % cC);
    size_t s = i / cC;
    int fx = (int)(s % Sf); size_t t = s / Sf;
    int fy = (int)(t % Sf); t /= Sf;
    int fz = (int)(t % Sf); int b = (int)(t / Sf);
    int jz[2], jy[2], jx[2];
    int nz = upcand(fz, Sc, jz), ny = upcand(fy, Sc, jy), nx = upcand(fx, Sc, jx);
    float sum = 0.f;
    for (int a = 0; a < nz; ++a)
        for (int d = 0; d < ny; ++d)
            for (int e = 0; e < nx; ++e) {
                size_t ci = (((size_t)b * Sc + jz[a]) * Sc + jy[d]) * Sc + jx[e];
                sum += coarse[ci * csStride + c];
            }
    fine[s * fsStride + fOff + c] = sum * fm[s];
}

// ---------------- implicit-GEMM conv via WMMA bf16 ----------------
// in : bf16 [B, Di,Di,Di, Cin]  (dense rows)
// wgt: bf16 [taps][Cout][Cin]   (pre-transposed)
// out: fp32 [Msites, outStride] written at [outOff .. outOff+coutStore)
// 128 threads = 4 waves per block; each wave owns one 16-site M tile.
__global__ void __launch_bounds__(128)
k_conv_wmma(const unsigned short* __restrict__ in,
            const unsigned short* __restrict__ wgt,
            const float* __restrict__ mask,
            const float* __restrict__ bias,
            float* __restrict__ out,
            int B, int Di, int Do, int Cin, int Cout,
            int stride, int pad, int kd,
            int outStride, int outOff, int coutStore) {
    const int lane = threadIdx.x & 31;
    const int wave = threadIdx.x >> 5;
    const int mtile = blockIdx.x * 4 + wave;
    const int hl = lane >> 4;          // half-wave
    const int mrow = lane & 15;        // A-matrix row for this lane
    const int M = B * Do * Do * Do;
    const int site = mtile * 16 + mrow;
    const bool siteOk = site < M;
    int b = 0, zo = 0, yo = 0, xo = 0;
    {
        int t = siteOk ? site : 0;
        xo = t % Do; t /= Do;
        yo = t % Do; t /= Do;
        zo = t % Do; b = t / Do;
    }
    const int ncol = blockIdx.y * 16 + mrow;   // B col / D col
    v8f acc = {0.f, 0.f, 0.f, 0.f, 0.f, 0.f, 0.f, 0.f};
    const int nk = Cin >> 5;
    const int taps = kd * kd * kd;

    for (int tap = 0; tap < taps; ++tap) {
        int dz = tap / (kd * kd);
        int r2 = tap - dz * kd * kd;
        int dy = r2 / kd;
        int dx = r2 - dy * kd;
        int zi = zo * stride + dz - pad;
        int yi = yo * stride + dy - pad;
        int xi = xo * stride + dx - pad;
        bool v = siteOk && (unsigned)zi < (unsigned)Di && (unsigned)yi < (unsigned)Di &&
                 (unsigned)xi < (unsigned)Di;
        size_t inBase = v ? (((((size_t)b * Di + zi) * Di + yi) * Di + xi) * (size_t)Cin) : 0;
        const unsigned short* wb = wgt + ((size_t)tap * Cout + ncol) * (size_t)Cin;
        for (int kc = 0; kc < nk; ++kc) {
            Frag af, bf;
            if (v) {
                // A: lane row mrow, channels [kc*32+hl*8, +8) and [kc*32+16+hl*8, +8)
                const U4* p = (const U4*)(in + inBase + (kc << 5) + (hl << 3));
                af.q[0] = p[0];
                af.q[1] = p[2];
            } else {
                af.q[0] = U4{0u, 0u, 0u, 0u};
                af.q[1] = U4{0u, 0u, 0u, 0u};
            }
            {
                // B: col ncol, K = kc*32 + hl*8 + [0..7] and kc*32 + 16 + hl*8 + [0..7]
                const U4* q = (const U4*)(wb + (kc << 5) + (hl << 3));
                bf.q[0] = q[0];
                bf.q[1] = q[2];
            }
            acc = __builtin_amdgcn_wmma_f32_16x16x32_bf16(
                false, af.v, false, bf.v, (short)0, acc, false, false);
        }
    }

    // D layout: VGPR r holds M = r + 8*hl, col = lane&15
#pragma unroll
    for (int r = 0; r < 8; ++r) {
        int srow = r + (hl << 3);
        int ssite = mtile * 16 + srow;
        if (ssite < M && ncol < coutStore) {
            float v = acc[r];
            if (bias) v += bias[ncol];
            out[(size_t)ssite * outStride + outOff + ncol] = v * mask[ssite];
        }
    }
}

// ---------------- host orchestration ----------------

static inline size_t align256(size_t o) { return (o + 255) & ~(size_t)255; }

extern "C" void kernel_launch(void* const* d_in, const int* in_sizes, int n_in,
                              void* d_out, int out_size, void* d_ws, size_t ws_size,
                              hipStream_t stream) {
    (void)in_sizes; (void)out_size; (void)ws_size;
    if (n_in < 34) return;

    const int B = 4;
    const int S[5] = {32, 15, 7, 3, 1};
    const int P[5] = {32, 64, 96, 128, 160};
    const int CCh[5] = {480, 448, 384, 288, 160};
    int M[5];
    for (int l = 0; l < 5; ++l) M[l] = B * S[l] * S[l] * S[l];

    const float* x = (const float*)d_in[0];
    const int* mi = (const int*)d_in[1];
    const float* w_in = (const float*)d_in[2];
    const float* bw[5]; const float* bg[5]; const float* bb[5];
    for (int l = 0; l < 5; ++l) {
        bw[l] = (const float*)d_in[3 + l];
        bg[l] = (const float*)d_in[8 + l];
        bb[l] = (const float*)d_in[13 + l];
    }
    const float* dw[4]; const float* dg[4]; const float* db[4];
    for (int l = 0; l < 4; ++l) {
        dw[l] = (const float*)d_in[18 + l];
        dg[l] = (const float*)d_in[22 + l];
        db[l] = (const float*)d_in[26 + l];
    }
    const float* bnf_g = (const float*)d_in[30];
    const float* bnf_b = (const float*)d_in[31];
    const float* lw = (const float*)d_in[32];
    const float* lb = (const float*)d_in[33];
    float* outp = (float*)d_out;

    // ---- workspace carve ----
    size_t off = 0;
    char* W = (char*)d_ws;
    auto carve = [&](size_t bytes) -> void* {
        size_t o = align256(off);
        off = o + bytes;
        return (void*)(W + o);
    };
    float* mf[5];
    for (int l = 0; l < 5; ++l) mf[l] = (float*)carve((size_t)M[l] * 4);
    float* h[5];
    for (int l = 0; l < 5; ++l) h[l] = (float*)carve((size_t)M[l] * P[l] * 4);
    float* cc[5];
    for (int l = 0; l < 5; ++l) cc[l] = (float*)carve((size_t)M[l] * CCh[l] * 4);
    unsigned short* act[5];
    for (int l = 0; l < 5; ++l) act[l] = (unsigned short*)carve((size_t)M[l] * P[l] * 2);
    unsigned short* actF = (unsigned short*)carve((size_t)M[0] * 480 * 2);
    unsigned short* wbB[5];
    for (int l = 0; l < 5; ++l) wbB[l] = (unsigned short*)carve((size_t)27 * P[l] * P[l] * 2);
    unsigned short* wbD[4];
    for (int l = 0; l < 4; ++l) wbD[l] = (unsigned short*)carve((size_t)27 * P[l + 1] * P[l] * 2);
    unsigned short* wbL = (unsigned short*)carve((size_t)64 * 480 * 2);
    float* stats = (float*)carve(961 * 4);

    dim3 blk(256);
    auto g1 = [](size_t n) { return dim3((unsigned)((n + 255) / 256)); };
    auto cgridM = [](int Msites) { return (unsigned)(((Msites + 15) / 16 + 3) / 4); };

    // mask + weight prep
    k_maskf<<<g1(M[0]), blk, 0, stream>>>(mi, mf[0], M[0]);
    for (int l = 0; l < 5; ++l)
        k_wconv<<<g1((size_t)27 * P[l] * P[l]), blk, 0, stream>>>(bw[l], wbB[l], 27, P[l], P[l], P[l]);
    for (int l = 0; l < 4; ++l)
        k_wconv<<<g1((size_t)27 * P[l + 1] * P[l]), blk, 0, stream>>>(dw[l], wbD[l], 27, P[l], P[l + 1], P[l + 1]);
    k_wconv<<<g1((size_t)64 * 480), blk, 0, stream>>>(lw, wbL, 1, 480, 50, 64);

    // input submanifold conv (Cin=1, fp32 path)
    k_input_conv<<<g1((size_t)M[0] * 32), blk, 0, stream>>>(x, mf[0], w_in, h[0], B, 32, 32);

    // ---- down pass ----
    for (int l = 0; l < 5; ++l) {
        int C = P[l];
        size_t tot = (size_t)M[l] * C;
        int nst = 1 + 2 * C;
        int nb = M[l] < 2048 ? M[l] : 2048;

        // block BN+ReLU on pre-block input h[l]
        k_zero<<<g1(nst), blk, 0, stream>>>(stats, nst);
        k_bn_stats<<<nb, 256, 0, stream>>>(h[l], mf[l], stats, M[l], C, C);
        k_bn_relu<<<g1(tot), blk, 0, stream>>>(h[l], mf[l], stats, bg[l], bb[l], act[l], M[l], C, C);

        // submanifold conv -> first P[l] channels of concat buffer
        dim3 cgrid(cgridM(M[l]), C / 16);
        k_conv_wmma<<<cgrid, 128, 0, stream>>>(act[l], wbB[l], mf[l], nullptr, cc[l],
                                               B, S[l], S[l], C, C, 1, 1, 3, CCh[l], 0, C);

        if (l < 4) {
            // down BN+ReLU on x_l (lives strided inside cc[l])
            k_zero<<<g1(nst), blk, 0, stream>>>(stats, nst);
            k_bn_stats<<<nb, 256, 0, stream>>>(cc[l], mf[l], stats, M[l], C, CCh[l]);
            k_bn_relu<<<g1(tot), blk, 0, stream>>>(cc[l], mf[l], stats, dg[l], db[l], act[l], M[l], C, CCh[l]);

            // coarse mask, then stride-2 VALID down conv
            k_cmask<<<g1(M[l + 1]), blk, 0, stream>>>(mf[l], mf[l + 1], B, S[l], S[l + 1]);
            dim3 dgrid(cgridM(M[l + 1]), P[l + 1] / 16);
            k_conv_wmma<<<dgrid, 128, 0, stream>>>(act[l], wbD[l], mf[l + 1], nullptr, h[l + 1],
                                                   B, S[l], S[l + 1], P[l], P[l + 1], 2, 0, 3,
                                                   P[l + 1], 0, P[l + 1]);
        }
    }

    // ---- up pass: unpool coarse concat into fine concat tail ----
    for (int l = 3; l >= 0; --l) {
        size_t tot = (size_t)M[l] * CCh[l + 1];
        k_unpool<<<g1(tot), blk, 0, stream>>>(cc[l + 1], CCh[l + 1], CCh[l + 1],
                                              cc[l], CCh[l], P[l], mf[l], B, S[l + 1], S[l]);
    }

    // ---- final BN+ReLU over 480 channels ----
    {
        size_t tot = (size_t)M[0] * 480;
        int nb = M[0] < 2048 ? M[0] : 2048;
        k_zero<<<g1(961), blk, 0, stream>>>(stats, 961);
        k_bn_stats<<<nb, 256, 0, stream>>>(cc[0], mf[0], stats, M[0], 480, 480);
        k_bn_relu<<<g1(tot), blk, 0, stream>>>(cc[0], mf[0], stats, bnf_g, bnf_b, actF, M[0], 480, 480);
    }

    // ---- per-voxel linear 480->50 as 1x1x1 WMMA "conv" (Cout padded to 64) ----
    {
        dim3 lgrid(cgridM(M[0]), 4);
        k_conv_wmma<<<lgrid, 128, 0, stream>>>(actF, wbL, mf[0], lb, outp,
                                               B, 32, 32, 480, 64, 1, 0, 1, 50, 0, 50);
    }
}
